// fcn_81518479278816
// MI455X (gfx1250) — matrix operand order repaired and checked by reading, not compiled, not run on hardware
//
#include <hip/hip_runtime.h>
#include <hip/hip_bf16.h>

// ---------------------------------------------------------------------------
// Hypernetwork CNN for MI455X (gfx1250), compute path = v_wmma_f32_16x16x32_bf16.
// Activations: NHWC bf16 with a 4-pixel zero halo so all conv B-operand loads
// are unconditional aligned b128 loads. GEMM K-order k = (kh*5+kw)*CIN + ci.
// Weights are hypernetwork-generated directly into the WMMA A-fragment layout.
// Conv K-loop is manually double-buffered (no cross-iteration register copies)
// and the tap decode is wave-uniform so it scalarizes.
// ---------------------------------------------------------------------------

#define BATCH   32
#define NA_DIM  32
#define HALO    4

typedef __attribute__((ext_vector_type(16))) __bf16 bf16x16;
typedef __attribute__((ext_vector_type(8)))  __bf16 bf16x8;
typedef __attribute__((ext_vector_type(8)))  float  v8f;

// K-index held by (lane, half j) of a 16-bit 16x32 fragment (ISA 7.12.2):
// j<8 : k = base + j ; j>=8 : k = base + 16 + (j-8) ; base = (lane<16)?0:8
__device__ __forceinline__ int frag_k(int lane, int j) {
    int base = (lane < 16) ? 0 : 8;
    return (j < 8) ? (base + j) : (base + 16 + (j - 8));
}

// ---------------------------------------------------------------------------
// Weight generation: y = W @ a + b written as bf16 in A-fragment layout.
// Fragment K-order: k = tap*CIN + ci (tap = kh*5+kw); W-matrix row is
// co*CIN*25 + ci*25 + tap.  frag[((b*nCoT+cot)*KC+kc)*512 + lane*16 + j]
// k >= CIN*25 gets an exact 0 (tail chunks contribute nothing).
// ---------------------------------------------------------------------------
__global__ void genfrag_kernel(const float* __restrict__ Wm,
                               const float* __restrict__ bias,
                               const float* __restrict__ action,
                               __bf16* __restrict__ frag,
                               int CIN, int nCoT, int KC) {
    int tid = blockIdx.x * blockDim.x + threadIdx.x;
    int total = BATCH * nCoT * KC * 512;
    if (tid >= total) return;
    int j    = tid & 15;
    int lane = (tid >> 4) & 31;
    int rest = tid >> 9;
    int kc   = rest % KC;   rest /= KC;
    int cot  = rest % nCoT;
    int b    = rest / nCoT;

    int co = cot * 16 + (lane & 15);
    int k  = kc * 32 + frag_k(lane, j);
    int K  = CIN * 25;
    float val = 0.f;
    if (k < K) {
        int tap = k / CIN, ci = k % CIN;
        int row = co * K + ci * 25 + tap;
        const float* wr = Wm + (size_t)row * NA_DIM;
        const float* av = action + b * NA_DIM;
        #pragma unroll
        for (int n = 0; n < NA_DIM; ++n) val += wr[n] * av[n];
        val += bias[row];
    }
    frag[tid] = (__bf16)val;
}

__global__ void gen_y7_kernel(const float* __restrict__ Wm,
                              const float* __restrict__ bias,
                              const float* __restrict__ action,
                              float* __restrict__ y7) {
    int tid = blockIdx.x * blockDim.x + threadIdx.x;
    if (tid >= BATCH * 512) return;
    int r = tid & 511, b = tid >> 9;
    const float* wr = Wm + (size_t)r * NA_DIM;
    const float* av = action + b * NA_DIM;
    float val = 0.f;
    #pragma unroll
    for (int n = 0; n < NA_DIM; ++n) val += wr[n] * av[n];
    y7[tid] = val + bias[r];
}

// ---------------------------------------------------------------------------
// Zero the 4-pixel halo of a padded NHWC bf16 buffer (per launch, cheap).
// ---------------------------------------------------------------------------
__global__ void halo_zero_kernel(__bf16* __restrict__ buf, int Hp, int Wp, int C) {
    int PH = 8 * Wp + 8 * (Hp - 8);          // halo pixels per sample
    size_t idx = (size_t)blockIdx.x * blockDim.x + threadIdx.x;
    size_t total = (size_t)BATCH * PH * C;
    if (idx >= total) return;
    int c  = (int)(idx % C);
    size_t rest = idx / C;
    int hp = (int)(rest % PH);
    int b  = (int)(rest / PH);
    int row, col;
    if (hp < 4 * Wp)            { row = hp / Wp;                 col = hp % Wp; }
    else if (hp < 8 * Wp)       { int q = hp - 4 * Wp;
                                  row = Hp - 4 + q / Wp;         col = q % Wp; }
    else                        { int q = hp - 8 * Wp;
                                  row = 4 + q / 8; int c8 = q % 8;
                                  col = (c8 < 4) ? c8 : Wp - 8 + c8; }
    buf[((size_t)b * Hp * Wp + (size_t)row * Wp + col) * C + c] = (__bf16)0.f;
}

// ---------------------------------------------------------------------------
// BN stats for the NCHW f32 network input: one block per (b,c)
// ---------------------------------------------------------------------------
__global__ void bn_stats_nchw(const float* __restrict__ z,
                              float* __restrict__ stats, int HW) {
    int bc = blockIdx.x;
    const float* p = z + (size_t)bc * HW;
    float s = 0.f, s2 = 0.f;
    for (int i = threadIdx.x; i < HW; i += blockDim.x) {
        float v = p[i]; s += v; s2 += v * v;
    }
    __shared__ float sh[256], sh2[256];
    sh[threadIdx.x] = s; sh2[threadIdx.x] = s2;
    __syncthreads();
    for (int o = 128; o > 0; o >>= 1) {
        if (threadIdx.x < o) {
            sh[threadIdx.x]  += sh[threadIdx.x + o];
            sh2[threadIdx.x] += sh2[threadIdx.x + o];
        }
        __syncthreads();
    }
    if (threadIdx.x == 0) {
        float m   = sh[0] / (float)HW;
        float var = sh2[0] / (float)HW - m * m;
        stats[2 * bc]     = m;
        stats[2 * bc + 1] = rsqrtf(var + 1e-5f);
    }
}

// BN0: read NCHW f32 x, write padded NHWC bf16 (C=8, 256->264 padded).
__global__ void bn0_apply_kernel(const float* __restrict__ x,
                                 const float* __restrict__ stats,
                                 const float* __restrict__ g,
                                 const float* __restrict__ be,
                                 __bf16* __restrict__ out) {
    size_t idx = (size_t)blockIdx.x * blockDim.x + threadIdx.x;
    if (idx >= (size_t)BATCH * 65536) return;
    int b = (int)(idx >> 16), p = (int)(idx & 65535);
    int y = p >> 8, xcol = p & 255;
    const int Wp = 264;
    const float* xb = x + (size_t)b * 8 * 65536 + p;
    __bf16* ob = out + ((size_t)b * 264 * 264 + (size_t)(y + HALO) * Wp + (xcol + HALO)) * 8;
    #pragma unroll
    for (int c = 0; c < 8; ++c) {
        float m = stats[2 * (b * 8 + c)], inv = stats[2 * (b * 8 + c) + 1];
        ob[c] = (__bf16)((xb[(size_t)c * 65536] - m) * inv * g[c] + be[c]);
    }
}

// ---------------------------------------------------------------------------
// BN stats over unpadded NHWC f32 (two-pass, coalesced channel-interleaved).
// ---------------------------------------------------------------------------
__global__ void bn_stats_part_nhwc(const float* __restrict__ z,
                                   float* __restrict__ part,
                                   int C, int HW, int NS) {
    int b = blockIdx.x / NS, sl = blockIdx.x % NS;
    int perSlice = HW / NS;
    const float* base = z + ((size_t)b * HW + (size_t)sl * perSlice) * C;
    int n = perSlice * C;
    int t = threadIdx.x;
    float s = 0.f, s2 = 0.f;
    for (int i = t; i < n; i += 256) {       // i % C == t % C (256 % C == 0)
        float v = base[i]; s += v; s2 += v * v;
    }
    __shared__ float sh[256], sh2[256];
    sh[t] = s; sh2[t] = s2;
    __syncthreads();
    for (int o = 128; o >= C; o >>= 1) {
        if (t < o) { sh[t] += sh[t + o]; sh2[t] += sh2[t + o]; }
        __syncthreads();
    }
    if (t < C) {
        float* pp = part + ((size_t)(b * NS + sl) * C + t) * 2;
        pp[0] = sh[t]; pp[1] = sh2[t];
    }
}

__global__ void bn_stats_combine(const float* __restrict__ part,
                                 float* __restrict__ stats,
                                 int C, int NS, int HW) {
    int tid = blockIdx.x * blockDim.x + threadIdx.x;
    if (tid >= BATCH * C) return;
    int b = tid / C, c = tid % C;
    float s = 0.f, s2 = 0.f;
    for (int sl = 0; sl < NS; ++sl) {
        const float* pp = part + ((size_t)(b * NS + sl) * C + c) * 2;
        s += pp[0]; s2 += pp[1];
    }
    float m   = s / (float)HW;
    float var = s2 / (float)HW - m * m;
    stats[2 * tid]     = m;
    stats[2 * tid + 1] = rsqrtf(var + 1e-5f);
}

// BN apply: unpadded NHWC f32 -> padded NHWC bf16 (+optional ReLU).
__global__ void bn_apply_nhwc(const float* __restrict__ z,
                              const float* __restrict__ stats,
                              const float* __restrict__ g,
                              const float* __restrict__ be,
                              __bf16* __restrict__ out,
                              int C, int HW, int W, int WSH, int Wp, int relu) {
    size_t idx = (size_t)blockIdx.x * blockDim.x + threadIdx.x;
    size_t total = (size_t)BATCH * C * HW;
    if (idx >= total) return;
    int c = (int)(idx % C);
    size_t pix = idx / C;
    int p = (int)(pix % HW);
    int b = (int)(pix / HW);
    int bc = b * C + c;
    float m = stats[2 * bc], inv = stats[2 * bc + 1];
    float v = (z[idx] - m) * inv * g[c] + be[c];
    if (relu) v = fmaxf(v, 0.f);
    int y = p >> WSH, xcol = p & (W - 1);
    int Hp = (HW >> WSH) + 2 * HALO;
    out[((size_t)b * Hp * Wp + (size_t)(y + HALO) * Wp + (xcol + HALO)) * C + c] = (__bf16)v;
}

// 2x2 avg pool: padded NHWC bf16 (264-geom) -> padded NHWC bf16 (136-geom)
__global__ void pool2_nhwc(const __bf16* __restrict__ in,
                           __bf16* __restrict__ out, int C, int HO) {
    size_t idx = (size_t)blockIdx.x * blockDim.x + threadIdx.x;
    size_t total = (size_t)BATCH * C * HO * HO;
    if (idx >= total) return;
    int c  = (int)(idx % C);
    size_t pix = idx / C;
    int ox = (int)(pix % HO);
    int oy = (int)((pix / HO) % HO);
    int b  = (int)(pix / ((size_t)HO * HO));
    const int WpI = 2 * HO + 8, WpO = HO + 8;
    const __bf16* p = in + ((size_t)b * WpI * WpI) * C + c;
    size_t r0 = ((size_t)(2 * oy + HALO) * WpI + (2 * ox + HALO)) * C;
    size_t r1 = r0 + (size_t)WpI * C;
    float s = (float)p[r0] + (float)p[r0 + C] + (float)p[r1] + (float)p[r1 + C];
    out[((size_t)b * WpO * WpO + (size_t)(oy + HALO) * WpO + (ox + HALO)) * C + c] =
        (__bf16)(0.25f * s);
}

// ---------------------------------------------------------------------------
// 5x5 conv (stride 1, pad=2*DIL) as implicit GEMM on WMMA.
// Block = 4 waves; each wave owns NTIL 16-pixel N-tiles in one row and NCOT
// 16-channel M-tiles.  Padded NHWC bf16 in, unpadded NHWC f32 out.
// Unconditional loads; tap decode is wave-uniform (scalarizes) for CIN>=16;
// manual 2x unrolled K-loop with ping-pong B fragments (no register copies).
// ---------------------------------------------------------------------------
template <int NCOT, int NTIL, int CSH, int DIL>
__global__ __launch_bounds__(128)
void conv5x5_wmma_kernel(const __bf16* __restrict__ in,
                         const __bf16* __restrict__ wfrag,
                         float* __restrict__ out,
                         int H, int W, int KC) {
    const int CIN = 1 << CSH;
    const int Hp = H + 2 * HALO, Wp = W + 2 * HALO;
    const int OFF = HALO - 2 * DIL;              // >= 0
    const int lane = threadIdx.x & 31;
    const int wv   = threadIdx.x >> 5;
    const int pixPerBlock = 64 * NTIL;
    const int tilesPer = (H * W) / pixPerBlock;
    const int b  = blockIdx.x / tilesPer;
    const int t  = blockIdx.x % tilesPer;
    const int pbase = t * pixPerBlock + wv * (16 * NTIL);
    const int oy  = pbase / W;                   // wave-uniform row
    const int oxb = (pbase - oy * W) + (lane & 15);
    const int base = (lane < 16) ? 0 : 8;
    const int sW = DIL * Wp;
    (void)Hp;

    const __bf16* inb = in + (size_t)b * Hp * Wp * CIN;
    const __bf16* wb  = wfrag + (size_t)b * NCOT * KC * 512 + lane * 16;

    // per-tile base offsets (halfs) incl. the lane ci part (base)
    int tbb[NTIL];
    #pragma unroll
    for (int nt = 0; nt < NTIL; ++nt)
        tbb[nt] = ((((oy + OFF) * Wp) + (oxb + nt * 16 + OFF)) << CSH) + base;

    // tap -> pixel offset (halfs); clamp keeps K-tail addresses in the map
    auto tapoff = [&](int tap) -> int {
        tap = min(tap, 24);
        int kh = tap / 5, kw = tap - kh * 5;
        return (kh * sW + kw * DIL) << CSH;
    };
    // offsets of the two 8-half runs of chunk kc (excluding tbb)
    auto chunkoffs = [&](int kc, int& olo, int& ohi) {
        if constexpr (CSH >= 4) {
            // tap and scalar ci part are wave-uniform (base never carries)
            int kslo = kc * 32, kshi = kslo + 16;
            olo = tapoff(kslo >> CSH) + (kslo & (CIN - 1));
            ohi = tapoff(kshi >> CSH) + (kshi & (CIN - 1));
        } else {                                  // CIN==8: tap differs per half-wave
            int t0 = tapoff(4 * kc), t1 = tapoff(4 * kc + 1);
            int t2 = tapoff(4 * kc + 2), t3 = tapoff(4 * kc + 3);
            bool hi = lane >= 16;
            olo = hi ? t1 - 8 : t0;
            ohi = hi ? t3 - 8 : t2;
        }
    };
    auto loadB = [&](int olo, int ohi, int nt) -> bf16x16 {
        bf16x8 lo = *reinterpret_cast<const bf16x8*>(inb + tbb[nt] + olo);
        bf16x8 hi = *reinterpret_cast<const bf16x8*>(inb + tbb[nt] + ohi);
        return __builtin_shufflevector(lo, hi, 0, 1, 2, 3, 4, 5, 6, 7,
                                       8, 9, 10, 11, 12, 13, 14, 15);
    };

    v8f zero = {};
    v8f acc[NCOT][NTIL];
    #pragma unroll
    for (int c = 0; c < NCOT; ++c)
        #pragma unroll
        for (int nt = 0; nt < NTIL; ++nt) acc[c][nt] = zero;

    bf16x16 a0[NCOT], a1[NCOT];
    bf16x16 b0[NTIL], b1[NTIL];
    {
        int olo, ohi;
        chunkoffs(0, olo, ohi);
        #pragma unroll
        for (int nt = 0; nt < NTIL; ++nt) b0[nt] = loadB(olo, ohi, nt);
    }

    int kc = 0;
    for (; kc + 1 < KC; kc += 2) {
        // ---- even chunk: consume b0, prefetch b1 ----
        #pragma unroll
        for (int c = 0; c < NCOT; ++c)
            a0[c] = *reinterpret_cast<const bf16x16*>(wb + (size_t)(c * KC + kc) * 512);
        {
            int olo, ohi; chunkoffs(kc + 1, olo, ohi);
            #pragma unroll
            for (int nt = 0; nt < NTIL; ++nt) b1[nt] = loadB(olo, ohi, nt);
        }
        #pragma unroll
        for (int c = 0; c < NCOT; ++c)
            #pragma unroll
            for (int nt = 0; nt < NTIL; ++nt)
                acc[c][nt] = __builtin_amdgcn_wmma_f32_16x16x32_bf16(
                    false, a0[c], false, b0[nt], (short)0, acc[c][nt], false, false);
        // ---- odd chunk: consume b1, prefetch b0 ----
        #pragma unroll
        for (int c = 0; c < NCOT; ++c)
            a1[c] = *reinterpret_cast<const bf16x16*>(wb + (size_t)(c * KC + kc + 1) * 512);
        {
            int olo, ohi; chunkoffs(kc + 2, olo, ohi);   // kc+2<=KC, clamp-safe
            #pragma unroll
            for (int nt = 0; nt < NTIL; ++nt) b0[nt] = loadB(olo, ohi, nt);
        }
        #pragma unroll
        for (int c = 0; c < NCOT; ++c)
            #pragma unroll
            for (int nt = 0; nt < NTIL; ++nt)
                acc[c][nt] = __builtin_amdgcn_wmma_f32_16x16x32_bf16(
                    false, a1[c], false, b1[nt], (short)0, acc[c][nt], false, false);
    }
    if (kc < KC) {                                       // KC odd: last chunk
        #pragma unroll
        for (int c = 0; c < NCOT; ++c)
            a0[c] = *reinterpret_cast<const bf16x16*>(wb + (size_t)(c * KC + kc) * 512);
        #pragma unroll
        for (int c = 0; c < NCOT; ++c)
            #pragma unroll
            for (int nt = 0; nt < NTIL; ++nt)
                acc[c][nt] = __builtin_amdgcn_wmma_f32_16x16x32_bf16(
                    false, a0[c], false, b0[nt], (short)0, acc[c][nt], false, false);
    }

    // C/D layout: N = lane&15; VGPR r -> channel (c*16 + mo + r), r contiguous
    const int n  = lane & 15;
    const int mo = (lane < 16) ? 0 : 8;
    const int COUT = NCOT * 16;
    #pragma unroll
    for (int nt = 0; nt < NTIL; ++nt) {
        float* op = out + ((size_t)b * H * W + (pbase + nt * 16 + n)) * COUT;
        #pragma unroll
        for (int c = 0; c < NCOT; ++c)
            *reinterpret_cast<v8f*>(op + c * 16 + mo) = acc[c][nt];
    }
}

// ---------------------------------------------------------------------------
// 1x1 conv: padded NHWC bf16 (136-geom, C=64) -> unpadded NHWC f32 (C=8)
// ---------------------------------------------------------------------------
__global__ void conv1x1_kernel(const __bf16* __restrict__ in,
                               const float* __restrict__ y7,
                               float* __restrict__ out) {
    int idx = blockIdx.x * blockDim.x + threadIdx.x;
    if (idx >= BATCH * 16384) return;
    int b = idx >> 14, p = idx & 16383;
    int y = p >> 7, xcol = p & 127;
    const int Wp = 136;
    const __bf16* ib = in + ((size_t)b * Wp * Wp + (size_t)(y + HALO) * Wp + (xcol + HALO)) * 64;
    const float*  w  = y7 + b * 512;
    float xin[64];
    #pragma unroll
    for (int v = 0; v < 8; ++v) {
        bf16x8 c8 = *reinterpret_cast<const bf16x8*>(ib + v * 8);
        #pragma unroll
        for (int j = 0; j < 8; ++j) xin[v * 8 + j] = (float)c8[j];
    }
    float* op = out + (size_t)idx * 8;
    #pragma unroll
    for (int co = 0; co < 8; ++co) {
        float s = 0.f;
        #pragma unroll
        for (int ci = 0; ci < 64; ++ci) s += w[co * 64 + ci] * xin[ci];
        op[co] = s;
    }
}

// ---------------------------------------------------------------------------
// bilinear upsample x2 (align_corners) + channel softmax; c7 NHWC f32 in,
// NCHW f32 out (reference output layout).
// ---------------------------------------------------------------------------
__global__ void up_softmax_kernel(const float* __restrict__ c7,
                                  float* __restrict__ out) {
    size_t idx = (size_t)blockIdx.x * blockDim.x + threadIdx.x;
    if (idx >= (size_t)BATCH * 65536) return;
    int b  = (int)(idx >> 16);
    int p  = (int)(idx & 65535);
    int oy = p >> 8, ox = p & 255;
    const float scale = 127.0f / 255.0f;     // (n-1)/(2n-1), n=128
    float sy = oy * scale, sx = ox * scale;
    int y0 = (int)floorf(sy); int y1 = min(y0 + 1, 127); float fy = sy - y0;
    int x0 = (int)floorf(sx); int x1 = min(x0 + 1, 127); float fx = sx - x0;

    const float* basep = c7 + (size_t)b * 16384 * 8;
    const float* r00 = basep + (size_t)(y0 * 128 + x0) * 8;
    const float* r01 = basep + (size_t)(y0 * 128 + x1) * 8;
    const float* r10 = basep + (size_t)(y1 * 128 + x0) * 8;
    const float* r11 = basep + (size_t)(y1 * 128 + x1) * 8;

    float v[8], m = -3.4e38f;
    #pragma unroll
    for (int c = 0; c < 8; ++c) {
        v[c] = (r00[c] * (1.f - fy) + r10[c] * fy) * (1.f - fx)
             + (r01[c] * (1.f - fy) + r11[c] * fy) * fx;
        m = fmaxf(m, v[c]);
    }
    float s = 0.f;
    #pragma unroll
    for (int c = 0; c < 8; ++c) { v[c] = expf(v[c] - m); s += v[c]; }
    float inv = 1.f / s;
    #pragma unroll
    for (int c = 0; c < 8; ++c)
        out[((size_t)b * 8 + c) * 65536 + p] = v[c] * inv;
}

// ---------------------------------------------------------------------------
extern "C" void kernel_launch(void* const* d_in, const int* in_sizes, int n_in,
                              void* d_out, int out_size, void* d_ws, size_t ws_size,
                              hipStream_t stream) {
    const float* x      = (const float*)d_in[0];
    const float* action = (const float*)d_in[1];
    const float* Wl[7]  = {(const float*)d_in[2],  (const float*)d_in[4],
                           (const float*)d_in[6],  (const float*)d_in[8],
                           (const float*)d_in[10], (const float*)d_in[12],
                           (const float*)d_in[14]};
    const float* bl[7]  = {(const float*)d_in[3],  (const float*)d_in[5],
                           (const float*)d_in[7],  (const float*)d_in[9],
                           (const float*)d_in[11], (const float*)d_in[13],
                           (const float*)d_in[15]};
    const float* g[7], *be[7];
    for (int i = 0; i < 7; ++i) {
        g[i]  = (const float*)d_in[16 + 2 * i];
        be[i] = (const float*)d_in[17 + 2 * i];
    }
    float* outp = (float*)d_out;

    // ---- workspace layout (256B aligned) ----
    char* ws = (char*)d_ws;
    size_t off = 0;
    auto alloc = [&](size_t bytes) {
        size_t r = off; off += (bytes + 255) & ~(size_t)255; return r;
    };
    const int CINs[6]  = {8, 16, 16, 32, 32, 64};
    const int NCOTs[6] = {1, 1, 2, 2, 4, 4};
    const int KCs[6]   = {7, 13, 13, 25, 25, 50};
    __bf16* wf[6];
    for (int l = 0; l < 6; ++l)
        wf[l] = (__bf16*)(ws + alloc((size_t)BATCH * NCOTs[l] * KCs[l] * 512 * 2));
    float*  y7    = (float*)(ws + alloc((size_t)BATCH * 512 * 4));
    float*  fbuf  = (float*)(ws + alloc((size_t)BATCH * 16 * 65536 * 4));       // NHWC f32
    const size_t ABUF = (size_t)BATCH * 64 * 136 * 136;                          // max padded
    __bf16* abuf0 = (__bf16*)(ws + alloc(ABUF * 2));
    __bf16* abuf1 = (__bf16*)(ws + alloc(ABUF * 2));
    float*  stats = (float*)(ws + alloc((size_t)BATCH * 64 * 2 * 4));
    float*  part  = (float*)(ws + alloc((size_t)BATCH * 16 * 64 * 2 * 4));
    float*  c7buf = (float*)(ws + alloc((size_t)BATCH * 16384 * 8 * 4));        // NHWC f32
    (void)ws_size; (void)in_sizes; (void)n_in; (void)out_size;

    auto cdiv = [](size_t a, size_t b) { return (unsigned)((a + b - 1) / b); };
    const int NS = 16;
    auto halo = [&](__bf16* buf, int Hp, int Wp, int C) {
        size_t tot = (size_t)BATCH * (8 * Wp + 8 * (Hp - 8)) * C;
        halo_zero_kernel<<<cdiv(tot, 256), 256, 0, stream>>>(buf, Hp, Wp, C);
    };
    auto bn_nhwc = [&](const float* z, int C, int HW, int W, int WSH, int Wp,
                       const float* gg, const float* bb, __bf16* dst, int relu) {
        bn_stats_part_nhwc<<<BATCH * NS, 256, 0, stream>>>(z, part, C, HW, NS);
        bn_stats_combine<<<cdiv(BATCH * C, 256), 256, 0, stream>>>(part, stats, C, NS, HW);
        bn_apply_nhwc<<<cdiv((size_t)BATCH * C * HW, 256), 256, 0, stream>>>(
            z, stats, gg, bb, dst, C, HW, W, WSH, Wp, relu);
    };

    // ---- hypernetwork weights ----
    for (int l = 0; l < 6; ++l) {
        size_t tot = (size_t)BATCH * NCOTs[l] * KCs[l] * 512;
        genfrag_kernel<<<cdiv(tot, 256), 256, 0, stream>>>(
            Wl[l], bl[l], action, wf[l], CINs[l], NCOTs[l], KCs[l]);
    }
    gen_y7_kernel<<<cdiv(BATCH * 512, 256), 256, 0, stream>>>(Wl[6], bl[6], action, y7);

    // ---- BN0 on raw NCHW input -> padded NHWC bf16 ----
    bn_stats_nchw<<<BATCH * 8, 256, 0, stream>>>(x, stats, 65536);
    halo(abuf0, 264, 264, 8);
    bn0_apply_kernel<<<cdiv((size_t)BATCH * 65536, 256), 256, 0, stream>>>(
        x, stats, g[0], be[0], abuf0);

    // ---- L1: conv 8->16 dil2 @256; BN+ReLU ----
    conv5x5_wmma_kernel<1, 2, 3, 2><<<BATCH * (65536 / 128), 128, 0, stream>>>(
        abuf0, wf[0], fbuf, 256, 256, 7);
    halo(abuf1, 264, 264, 16);
    bn_nhwc(fbuf, 16, 65536, 256, 8, 264, g[1], be[1], abuf1, 1);

    // ---- L2: conv 16->16 dil1 @256; BN; pool -> 128 ----
    conv5x5_wmma_kernel<1, 2, 4, 1><<<BATCH * (65536 / 128), 128, 0, stream>>>(
        abuf1, wf[1], fbuf, 256, 256, 13);
    halo(abuf0, 264, 264, 16);
    bn_nhwc(fbuf, 16, 65536, 256, 8, 264, g[2], be[2], abuf0, 0);
    halo(abuf1, 136, 136, 16);
    pool2_nhwc<<<cdiv((size_t)BATCH * 16 * 128 * 128, 256), 256, 0, stream>>>(
        abuf0, abuf1, 16, 128);

    // ---- L3: conv 16->32 dil2 @128; BN+ReLU ----
    conv5x5_wmma_kernel<2, 1, 4, 2><<<BATCH * (16384 / 64), 128, 0, stream>>>(
        abuf1, wf[2], fbuf, 128, 128, 13);
    halo(abuf0, 136, 136, 32);
    bn_nhwc(fbuf, 32, 16384, 128, 7, 136, g[3], be[3], abuf0, 1);

    // ---- L4: conv 32->32 dil1 @128; BN ----
    conv5x5_wmma_kernel<2, 1, 5, 1><<<BATCH * (16384 / 64), 128, 0, stream>>>(
        abuf0, wf[3], fbuf, 128, 128, 25);
    halo(abuf1, 136, 136, 32);
    bn_nhwc(fbuf, 32, 16384, 128, 7, 136, g[4], be[4], abuf1, 0);

    // ---- L5: conv 32->64 dil2 @128; BN+ReLU ----
    conv5x5_wmma_kernel<4, 1, 5, 2><<<BATCH * (16384 / 64), 128, 0, stream>>>(
        abuf1, wf[4], fbuf, 128, 128, 25);
    halo(abuf0, 136, 136, 64);
    bn_nhwc(fbuf, 64, 16384, 128, 7, 136, g[5], be[5], abuf0, 1);

    // ---- L6: conv 64->64 dil1 @128; BN ----
    conv5x5_wmma_kernel<4, 1, 6, 1><<<BATCH * (16384 / 64), 128, 0, stream>>>(
        abuf0, wf[5], fbuf, 128, 128, 50);
    halo(abuf1, 136, 136, 64);
    bn_nhwc(fbuf, 64, 16384, 128, 7, 136, g[6], be[6], abuf1, 0);

    // ---- 1x1 conv -> bilinear up x2 + softmax ----
    conv1x1_kernel<<<cdiv((size_t)BATCH * 16384, 256), 256, 0, stream>>>(
        abuf1, y7, c7buf);
    up_softmax_kernel<<<cdiv((size_t)BATCH * 65536, 256), 256, 0, stream>>>(
        c7buf, outp);
}
